// ExchangeBlock_317827580343
// MI455X (gfx1250) — compile-verified, matrix operand order checked by reading, not run on hardware
//
#include <hip/hip_runtime.h>

// ---------------- Types ----------------
typedef __bf16 bf16;
typedef __attribute__((ext_vector_type(16))) __bf16 v16bf;
typedef __attribute__((ext_vector_type(8)))  float  v8f;

// ---------------- Problem constants ----------------
#define F_DIM   132
#define DIN     136          // F + 4 degree slots
#define MTOT    24           // 3+5+7+9
#define KPAD    160          // DIN padded to 5 chunks of 32
#define NCHUNK  5            // K chunks of 32
#define NTILE   9            // 144 output cols / 16 (DIN padded)
#define FRAG_ELEMS 512       // one 32x16 bf16 B fragment
#define WSLICE  (NCHUNK * NTILE * FRAG_ELEMS)   // 23040 bf16 per precision
#define WAVES   4
#define ROWS_PER_WAVE  16
#define ROWS_PER_BLOCK (WAVES * ROWS_PER_WAVE)  // 64

// K index held by (lane, element e) for both A and B fragments of
// v_wmma_f32_16x16x32_bf16 (ISA 7.12.2, 16-bit 16x32 layout):
//   lanes 0-15 : e 0..7 -> K = c*32 + e      ; e 8..15 -> K = c*32 + 16 + (e-8)
//   lanes 16-31: e 0..7 -> K = c*32 + 8 + e  ; e 8..15 -> K = c*32 + 24 + (e-8)
__device__ __forceinline__ int kmap(int lane, int e, int c) {
    int kbase = c * 32 + ((lane & 16) ? 8 : 0);
    return kbase + (e < 8 ? e : e + 8);
}

// ---------------- Prep: split W into bf16 hi/lo B-fragments ----------------
__global__ void prep_w_kernel(const float* __restrict__ W,
                              bf16* __restrict__ wh, bf16* __restrict__ wl) {
    int fid = blockIdx.x * blockDim.x + threadIdx.x;
    if (fid >= WSLICE) return;
    int ct   = fid / FRAG_ELEMS;       // which (chunk, tile)
    int idx  = fid % FRAG_ELEMS;       // element within fragment
    int c    = ct / NTILE;
    int t    = ct % NTILE;
    int lane = idx / 16;
    int e    = idx % 16;
    int k    = kmap(lane, e, c);
    int n    = t * 16 + (lane & 15);
    float w  = (k < DIN && n < DIN) ? W[k * DIN + n] : 0.0f;
    bf16  h  = (bf16)w;               // RNE truncation -> high part
    bf16  l  = (bf16)(w - (float)h);  // residual -> low part
    wh[fid] = h;
    wl[fid] = l;
}

// ---------------- Fused main kernel ----------------
__global__ void __launch_bounds__(128)
fused_exchange_kernel(const float* __restrict__ x,
                      const float* __restrict__ ev,
                      const float* __restrict__ b,
                      const bf16* __restrict__ wh,
                      const bf16* __restrict__ wl,
                      float* __restrict__ out_cx,
                      float* __restrict__ out_ev,
                      int n_rows)
{
    __shared__ float As  [WAVES][ROWS_PER_WAVE][KPAD];  // staged [x | contr | 0pad]
    __shared__ float evS [WAVES][ROWS_PER_WAVE][MTOT];  // staged ev tile
    __shared__ float cevS[WAVES][ROWS_PER_WAVE][4];     // y[:,132:136]

    const int lane = threadIdx.x & 31;
    const int w    = threadIdx.x >> 5;
    const int tileBase = blockIdx.x * ROWS_PER_BLOCK + w * ROWS_PER_WAVE;

    // ---- Stage x rows (coalesced float4); clamp tail rows ----
    #pragma unroll 2
    for (int i = lane; i < ROWS_PER_WAVE * (F_DIM / 4); i += 32) {
        int r  = i / (F_DIM / 4);
        int c4 = i % (F_DIM / 4);
        int grow = tileBase + r; if (grow >= n_rows) grow = n_rows - 1;
        float4 v = *(const float4*)(x + (size_t)grow * F_DIM + c4 * 4);
        *(float4*)(&As[w][r][c4 * 4]) = v;
    }
    // ---- Stage ev tile ----
    #pragma unroll 2
    for (int i = lane; i < ROWS_PER_WAVE * MTOT; i += 32) {
        int r = i / MTOT, m = i % MTOT;
        int grow = tileBase + r; if (grow >= n_rows) grow = n_rows - 1;
        evS[w][r][m] = ev[(size_t)grow * MTOT + m];
    }
    // ---- Per-row segment sums of ev^2 -> cols 132..135; zero pad 136..159 ----
    if (lane < ROWS_PER_WAVE) {
        int r = lane;
        float s0 = 0.f, s1 = 0.f, s2 = 0.f, s3 = 0.f;
        #pragma unroll
        for (int m = 0;  m < 3;  ++m) { float e = evS[w][r][m]; s0 += e * e; }
        #pragma unroll
        for (int m = 3;  m < 8;  ++m) { float e = evS[w][r][m]; s1 += e * e; }
        #pragma unroll
        for (int m = 8;  m < 15; ++m) { float e = evS[w][r][m]; s2 += e * e; }
        #pragma unroll
        for (int m = 15; m < 24; ++m) { float e = evS[w][r][m]; s3 += e * e; }
        As[w][r][F_DIM + 0] = s0;
        As[w][r][F_DIM + 1] = s1;
        As[w][r][F_DIM + 2] = s2;
        As[w][r][F_DIM + 3] = s3;
        #pragma unroll
        for (int k = DIN; k < KPAD; ++k) As[w][r][k] = 0.0f;
    }

    // ---- GEMM: y(16x144) = A(16x160) * W(160x144), bf16x3 split ----
    v8f acc[NTILE];
    #pragma unroll
    for (int t = 0; t < NTILE; ++t)
        #pragma unroll
        for (int j = 0; j < 8; ++j) acc[t][j] = 0.0f;

    const int m16 = lane & 15;
    #pragma unroll
    for (int c = 0; c < NCHUNK; ++c) {
        // Build A fragments (hi/lo split) from LDS
        v16bf Ah, Al;
        #pragma unroll
        for (int e = 0; e < 16; ++e) {
            int   k = kmap(lane, e, c);
            float v = As[w][m16][k];
            bf16  h = (bf16)v;
            Ah[e] = h;
            Al[e] = (bf16)(v - (float)h);
        }
        const v16bf* bh = (const v16bf*)(wh + (size_t)(c * NTILE) * FRAG_ELEMS + lane * 16);
        const v16bf* bl = (const v16bf*)(wl + (size_t)(c * NTILE) * FRAG_ELEMS + lane * 16);
        // Pass 1: Ah * Bh   (9 independent WMMAs -> no RAW hazard chain)
        #pragma unroll
        for (int t = 0; t < NTILE; ++t) {
            v16bf B = bh[t * (FRAG_ELEMS / 16)];
            acc[t] = __builtin_amdgcn_wmma_f32_16x16x32_bf16(
                false, Ah, false, B, (short)0, acc[t], false, false);
        }
        // Pass 2: Ah * Bl
        #pragma unroll
        for (int t = 0; t < NTILE; ++t) {
            v16bf B = bl[t * (FRAG_ELEMS / 16)];
            acc[t] = __builtin_amdgcn_wmma_f32_16x16x32_bf16(
                false, Ah, false, B, (short)0, acc[t], false, false);
        }
        // Pass 3: Al * Bh
        #pragma unroll
        for (int t = 0; t < NTILE; ++t) {
            v16bf B = bh[t * (FRAG_ELEMS / 16)];
            acc[t] = __builtin_amdgcn_wmma_f32_16x16x32_bf16(
                false, Al, false, B, (short)0, acc[t], false, false);
        }
    }

    // ---- Epilogue: bias + scatter cx, route cols 132..135 via LDS ----
    const int rhalf = (lane & 16) ? 8 : 0;   // C/D layout: lanes 16-31 hold M=r+8
    #pragma unroll
    for (int t = 0; t < NTILE; ++t) {
        int col = t * 16 + m16;
        float bias = (col < DIN) ? b[col] : 0.0f;
        #pragma unroll
        for (int r = 0; r < 8; ++r) {
            float yv = acc[t][r] + bias;
            int lr   = r + rhalf;
            int grow = tileBase + lr;
            if (col < F_DIM) {
                if (grow < n_rows) out_cx[(size_t)grow * F_DIM + col] = yv;
            } else if (col < DIN) {
                cevS[w][lr][col - F_DIM] = yv;
            }
        }
    }

    // ---- out2 = broadcast(cev over segments) * ev ----
    #pragma unroll 2
    for (int i = lane; i < ROWS_PER_WAVE * MTOT; i += 32) {
        int r = i / MTOT, m = i % MTOT;
        int grow = tileBase + r;
        if (grow < n_rows) {
            int s = (m < 3) ? 0 : (m < 8) ? 1 : (m < 15) ? 2 : 3;
            out_ev[(size_t)grow * MTOT + m] = cevS[w][r][s] * evS[w][r][m];
        }
    }
}

// ---------------- Launch ----------------
extern "C" void kernel_launch(void* const* d_in, const int* in_sizes, int n_in,
                              void* d_out, int out_size, void* d_ws, size_t ws_size,
                              hipStream_t stream) {
    const float* x  = (const float*)d_in[0];
    const float* ev = (const float*)d_in[1];
    const float* W  = (const float*)d_in[2];
    const float* b  = (const float*)d_in[3];
    const int n_rows = in_sizes[0] / F_DIM;

    float* out    = (float*)d_out;
    float* out_cx = out;                              // N x 132
    float* out_ev = out + (size_t)n_rows * F_DIM;     // N x 24

    bf16* wh = (bf16*)d_ws;
    bf16* wl = wh + WSLICE;

    prep_w_kernel<<<(WSLICE + 255) / 256, 256, 0, stream>>>(W, wh, wl);

    int blocks = (n_rows + ROWS_PER_BLOCK - 1) / ROWS_PER_BLOCK;
    fused_exchange_kernel<<<blocks, 128, 0, stream>>>(x, ev, b, wh, wl,
                                                      out_cx, out_ev, n_rows);
}
